// PointTransformerLayer_30640296689896
// MI455X (gfx1250) — matrix operand order repaired and checked by reading, not compiled, not run on hardware
//
#include <hip/hip_runtime.h>
#include <math.h>

// Problem constants (from reference)
#define BATCH   4
#define NPTS    16384
#define BN      65536          // BATCH * NPTS
#define IN_P    64
#define OUT_P   64
#define MID     16
#define RADIUS  8
#define S       16
#define EPS     1e-5f

typedef __attribute__((ext_vector_type(16))) _Float16 v16h;
typedef __attribute__((ext_vector_type(8)))  float    v8f;

// Issue one wave's 16x64 f32 tile (4 KB) as 8 per-lane async B128 DMAs:
// lane L copies bytes [L*16 + i*512, +16) of the tile, global -> LDS.
__device__ __forceinline__ void async_tile_load(const float* __restrict__ src,
                                                float* lds_tile, int lane)
{
    const unsigned ldsbase = (unsigned)(uintptr_t)lds_tile + (unsigned)lane * 16u;
    const char*    gbase   = (const char*)src + (size_t)lane * 16u;
    asm volatile("global_load_async_to_lds_b128 %0, %1, off offset:0"    :: "v"(ldsbase), "v"(gbase) : "memory");
    asm volatile("global_load_async_to_lds_b128 %0, %1, off offset:512"  :: "v"(ldsbase), "v"(gbase) : "memory");
    asm volatile("global_load_async_to_lds_b128 %0, %1, off offset:1024" :: "v"(ldsbase), "v"(gbase) : "memory");
    asm volatile("global_load_async_to_lds_b128 %0, %1, off offset:1536" :: "v"(ldsbase), "v"(gbase) : "memory");
    asm volatile("global_load_async_to_lds_b128 %0, %1, off offset:2048" :: "v"(ldsbase), "v"(gbase) : "memory");
    asm volatile("global_load_async_to_lds_b128 %0, %1, off offset:2560" :: "v"(ldsbase), "v"(gbase) : "memory");
    asm volatile("global_load_async_to_lds_b128 %0, %1, off offset:3072" :: "v"(ldsbase), "v"(gbase) : "memory");
    asm volatile("global_load_async_to_lds_b128 %0, %1, off offset:3584" :: "v"(ldsbase), "v"(gbase) : "memory");
}

// ---------------------------------------------------------------------------
// Stage 1: qkv projection GEMM via v_wmma_f32_16x16x32_f16
//   [65536 x 64] @ [64 x (16|16|64)] + bias  ->  q_ws, k_ws, v_ws (f32)
// One wave owns a 16-row M tile; 6 N-tiles (q, k, v0..v3); K=64 = 2 WMMA steps.
// B fragments (weights) live in registers for the whole grid-stride M loop.
// A tiles are double-buffered in wave-private LDS via async global->LDS DMA
// (ASYNCcnt), overlapping the next tile's fill with the current tile's WMMAs.
// ---------------------------------------------------------------------------
__global__ void __launch_bounds__(128) qkv_wmma_kernel(
    const float* __restrict__ feat,
    const float* __restrict__ Wq, const float* __restrict__ bq,
    const float* __restrict__ Wk, const float* __restrict__ bk,
    const float* __restrict__ Wv, const float* __restrict__ bv,
    float* __restrict__ q_ws, float* __restrict__ k_ws, float* __restrict__ v_ws)
{
    __shared__ __align__(16) float At[4][2][16 * 64];  // 2 buffers per wave

    const int lane = threadIdx.x & 31;
    const int wave = threadIdx.x >> 5;
    const int n15  = lane & 15;       // N (or M) index within 16
    const int hi   = lane >> 4;       // half-wave select
    const int hi8  = hi << 3;

    // ---- Preload 12 B fragments (6 N-tiles x 2 K-tiles) into VGPRs ----
    // B 32x16 f16 layout: lane L holds column N=L&15; half e holds K = e + 16*hi.
    v16h bf[6][2];
    #pragma unroll
    for (int t = 0; t < 6; ++t) {
        const float* W; int ld, cb;
        if (t == 0)      { W = Wq; ld = 16; cb = 0; }
        else if (t == 1) { W = Wk; ld = 16; cb = 0; }
        else             { W = Wv; ld = 64; cb = (t - 2) * 16; }
        #pragma unroll
        for (int kt = 0; kt < 2; ++kt) {
            const int kb = kt * 32 + hi * 16;
            #pragma unroll
            for (int e = 0; e < 16; ++e)
                bf[t][kt][e] = (_Float16)W[(kb + e) * ld + cb + n15];
        }
    }
    float bvals[6];
    bvals[0] = bq[n15];
    bvals[1] = bk[n15];
    #pragma unroll
    for (int j = 0; j < 4; ++j) bvals[2 + j] = bv[j * 16 + n15];

    const int wgid   = blockIdx.x * 4 + wave;
    const int nwaves = gridDim.x * 4;         // launched so 4096 % nwaves == 0
    const int trips  = (BN / 16) / nwaves;    // uniform across the whole block

    // Prologue: kick off tile 0 into buffer 0
    async_tile_load(feat + (size_t)wgid * (16 * 64), At[wave][0], lane);

    for (int tr = 0; tr < trips; ++tr) {
        const int mt = wgid + tr * nwaves;
        float* at = At[wave][tr & 1];

        if (tr + 1 < trips) {                 // uniform branch
            // ds reads of the buffer we are about to overwrite must be retired
            asm volatile("s_wait_dscnt 0" ::: "memory");
            async_tile_load(feat + (size_t)(mt + nwaves) * (16 * 64),
                            At[wave][(tr + 1) & 1], lane);
            // allow the 8 just-issued DMAs to stay in flight; wait for tile tr
            asm volatile("s_wait_asynccnt 8" ::: "memory");
        } else {
            asm volatile("s_wait_asynccnt 0" ::: "memory");
        }

        // A 16x32 f16 fragments: lane holds row M=n15; halves (e,e+1) -> K,K+1
        // with K = e + 16*hi.  Load K-pairs as float2 (b64 ds).
        v16h a0, a1;
        #pragma unroll
        for (int d = 0; d < 8; ++d) {
            const int kp = 2 * d + hi * 16;
            const float2 x0 = *(const float2*)&at[n15 * 64 + kp];
            const float2 x1 = *(const float2*)&at[n15 * 64 + 32 + kp];
            a0[2 * d]     = (_Float16)x0.x;
            a0[2 * d + 1] = (_Float16)x0.y;
            a1[2 * d]     = (_Float16)x1.x;
            a1[2 * d + 1] = (_Float16)x1.y;
        }

        const int rowbase = mt * 16 + hi8;   // C/D: VGPR r holds M = r + hi8
        #pragma unroll
        for (int t = 0; t < 6; ++t) {
            v8f acc;
            #pragma unroll
            for (int r = 0; r < 8; ++r) acc[r] = bvals[t];   // bias folded in
            acc = __builtin_amdgcn_wmma_f32_16x16x32_f16(false, a0, false, bf[t][0],
                                                         (short)0, acc, false, false);
            acc = __builtin_amdgcn_wmma_f32_16x16x32_f16(false, a1, false, bf[t][1],
                                                         (short)0, acc, false, false);
            #pragma unroll
            for (int r = 0; r < 8; ++r) {
                const int row = rowbase + r;
                if (t == 0)      q_ws[row * 16 + n15] = acc[r];
                else if (t == 1) k_ws[row * 16 + n15] = acc[r];
                else             v_ws[row * 64 + (t - 2) * 16 + n15] = acc[r];
            }
        }
    }
}

// ---------------------------------------------------------------------------
// Stage 2: ring-neighbor attention. One thread per point.
// out[c] = sum_n v_n[c]*w_n[c&7]
//        + Wp2[0][c]*sum_n(h0_n w_n[c&7]) + Wp2[1][c]*sum_n(h1_n w_n[c&7])
//        + bp2[c]*sum_n w_n[c&7]
// where h = relu(bn(t @ Wp1)) and w_n = softmax8(MLP(r_sum + k_n - q)).
// r_sum needs only column-group sums of Wp2/bp2 (precomputed into LDS).
// ---------------------------------------------------------------------------
__global__ void __launch_bounds__(256) pt_attn_kernel(
    const float* __restrict__ points,
    const float* __restrict__ q_ws, const float* __restrict__ k_ws,
    const float* __restrict__ v_ws,
    const float* __restrict__ Wp1,
    const float* __restrict__ p_g, const float* __restrict__ p_b,
    const float* __restrict__ p_m, const float* __restrict__ p_v,
    const float* __restrict__ Wp2, const float* __restrict__ bp2,
    const float* __restrict__ w1_g, const float* __restrict__ w1_b,
    const float* __restrict__ w1_m, const float* __restrict__ w1_v,
    const float* __restrict__ Ww1,
    const float* __restrict__ w2_g, const float* __restrict__ w2_b,
    const float* __restrict__ w2_m, const float* __restrict__ w2_v,
    const float* __restrict__ Ww2, const float* __restrict__ bw2,
    float* __restrict__ out)
{
    __shared__ float s_wp1[4];                       // Wp1 row-major [2][2]
    __shared__ float s_ps[2], s_pb[2];               // pos-BN scale/shift
    __shared__ float s_cs0[16], s_cs1[16], s_bps[16];// group-sums of Wp2/bp2
    __shared__ float s_w1s[16], s_w1b[16];           // w1-BN scale/shift
    __shared__ float s_ww1a[16], s_ww1b[16];         // Ww1 columns
    __shared__ float s_w2s[2], s_w2b[2];             // w2-BN scale/shift
    __shared__ float s_ww2[16], s_bw2[8];            // Ww2 [2][8], bw2
    __shared__ float s_wp2[128], s_bp2[64];          // raw Wp2 rows, bp2

    const int t = threadIdx.x;
    if (t < 4)   s_wp1[t] = Wp1[t];
    if (t < 128) s_wp2[t] = Wp2[t];
    if (t < 64)  s_bp2[t] = bp2[t];
    if (t < 2) {
        float sc = p_g[t] * rsqrtf(p_v[t] + EPS);
        s_ps[t] = sc; s_pb[t] = p_b[t] - p_m[t] * sc;
        float sc2 = w2_g[t] * rsqrtf(w2_v[t] + EPS);
        s_w2s[t] = sc2; s_w2b[t] = w2_b[t] - w2_m[t] * sc2;
    }
    if (t < 16) {
        s_cs0[t] = Wp2[t]      + Wp2[16 + t] + Wp2[32 + t]  + Wp2[48 + t];
        s_cs1[t] = Wp2[64 + t] + Wp2[80 + t] + Wp2[96 + t]  + Wp2[112 + t];
        s_bps[t] = bp2[t]      + bp2[16 + t] + bp2[32 + t]  + bp2[48 + t];
        float sc = w1_g[t] * rsqrtf(w1_v[t] + EPS);
        s_w1s[t] = sc; s_w1b[t] = w1_b[t] - w1_m[t] * sc;
        s_ww1a[t] = Ww1[t * 2 + 0];
        s_ww1b[t] = Ww1[t * 2 + 1];
        s_ww2[t]  = Ww2[t];
    }
    if (t < 8) s_bw2[t] = bw2[t];
    __syncthreads();

    const int i    = blockIdx.x * 256 + t;     // grid sized exactly to BN
    const int b    = i >> 14;
    const int n    = i & (NPTS - 1);
    const int base = b << 14;

    float q[16];
    const float4* qp = (const float4*)(q_ws + (size_t)i * 16);
    #pragma unroll
    for (int m4 = 0; m4 < 4; ++m4) {
        float4 qv = qp[m4];
        q[m4*4+0] = qv.x; q[m4*4+1] = qv.y; q[m4*4+2] = qv.z; q[m4*4+3] = qv.w;
    }
    const float px = points[2 * i], py = points[2 * i + 1];

    float acc[64];
    #pragma unroll
    for (int c = 0; c < 64; ++c) acc[c] = 0.f;
    float hw0[8], hw1[8], wsum[8];
    #pragma unroll
    for (int j = 0; j < 8; ++j) { hw0[j] = 0.f; hw1[j] = 0.f; wsum[j] = 0.f; }

    for (int s = 0; s < S; ++s) {              // ring: -8..-1, 1..8 (wrapped)
        const int off = (s < RADIUS) ? (s - RADIUS) : (s - RADIUS + 1);
        const int nb  = base + ((n + off) & (NPTS - 1));

        const float tx = points[2 * nb]     - px;
        const float ty = points[2 * nb + 1] - py;
        float h0 = fmaf(tx, s_wp1[0], ty * s_wp1[2]);
        float h1 = fmaf(tx, s_wp1[1], ty * s_wp1[3]);
        h0 = fmaxf(fmaf(h0, s_ps[0], s_pb[0]), 0.f);
        h1 = fmaxf(fmaf(h1, s_ps[1], s_pb[1]), 0.f);

        // attention-weight MLP: (r_sum + k - q) -> bn,relu -> 16x2 -> bn,relu -> 2x8
        float a0 = 0.f, a1 = 0.f;
        const float4* kp = (const float4*)(k_ws + (size_t)nb * 16);
        #pragma unroll
        for (int m4 = 0; m4 < 4; ++m4) {
            float4 kv = kp[m4];
            float kk[4] = {kv.x, kv.y, kv.z, kv.w};
            #pragma unroll
            for (int u = 0; u < 4; ++u) {
                const int m = m4 * 4 + u;
                float win = fmaf(h0, s_cs0[m], fmaf(h1, s_cs1[m], s_bps[m]))
                            + kk[u] - q[m];
                float act = fmaxf(fmaf(win, s_w1s[m], s_w1b[m]), 0.f);
                a0 = fmaf(act, s_ww1a[m], a0);
                a1 = fmaf(act, s_ww1b[m], a1);
            }
        }
        const float u0 = fmaxf(fmaf(a0, s_w2s[0], s_w2b[0]), 0.f);
        const float u1 = fmaxf(fmaf(a1, s_w2s[1], s_w2b[1]), 0.f);

        float wj[8];
        float wmax = -3.4e38f;
        #pragma unroll
        for (int j = 0; j < 8; ++j) {
            wj[j] = fmaf(u0, s_ww2[j], fmaf(u1, s_ww2[8 + j], s_bw2[j]));
            wmax = fmaxf(wmax, wj[j]);
        }
        float wden = 0.f;
        #pragma unroll
        for (int j = 0; j < 8; ++j) { wj[j] = __expf(wj[j] - wmax); wden += wj[j]; }
        const float inv = 1.f / wden;
        #pragma unroll
        for (int j = 0; j < 8; ++j) {
            wj[j] *= inv;
            hw0[j]  = fmaf(h0, wj[j], hw0[j]);
            hw1[j]  = fmaf(h1, wj[j], hw1[j]);
            wsum[j] += wj[j];
        }

        const float4* vp = (const float4*)(v_ws + (size_t)nb * 64);
        #pragma unroll
        for (int c4 = 0; c4 < 16; ++c4) {
            float4 vv = vp[c4];
            const int c = c4 * 4;
            acc[c + 0] = fmaf(vv.x, wj[(c + 0) & 7], acc[c + 0]);
            acc[c + 1] = fmaf(vv.y, wj[(c + 1) & 7], acc[c + 1]);
            acc[c + 2] = fmaf(vv.z, wj[(c + 2) & 7], acc[c + 2]);
            acc[c + 3] = fmaf(vv.w, wj[(c + 3) & 7], acc[c + 3]);
        }
    }

    // fold the factored positional-encoding contribution back in and store
    float4* op = (float4*)(out + (size_t)i * 64);
    #pragma unroll
    for (int c4 = 0; c4 < 16; ++c4) {
        const int c = c4 * 4;
        float4 o;
        o.x = acc[c+0] + fmaf(s_wp2[c+0], hw0[(c+0)&7],
                         fmaf(s_wp2[64+c+0], hw1[(c+0)&7], s_bp2[c+0] * wsum[(c+0)&7]));
        o.y = acc[c+1] + fmaf(s_wp2[c+1], hw0[(c+1)&7],
                         fmaf(s_wp2[64+c+1], hw1[(c+1)&7], s_bp2[c+1] * wsum[(c+1)&7]));
        o.z = acc[c+2] + fmaf(s_wp2[c+2], hw0[(c+2)&7],
                         fmaf(s_wp2[64+c+2], hw1[(c+2)&7], s_bp2[c+2] * wsum[(c+2)&7]));
        o.w = acc[c+3] + fmaf(s_wp2[c+3], hw0[(c+3)&7],
                         fmaf(s_wp2[64+c+3], hw1[(c+3)&7], s_bp2[c+3] * wsum[(c+3)&7]));
        op[c4] = o;
    }
}

// ---------------------------------------------------------------------------
extern "C" void kernel_launch(void* const* d_in, const int* in_sizes, int n_in,
                              void* d_out, int out_size, void* d_ws, size_t ws_size,
                              hipStream_t stream) {
    const float* points = (const float*)d_in[0];
    const float* feat   = (const float*)d_in[1];
    const float* Wq  = (const float*)d_in[2];
    const float* bq  = (const float*)d_in[3];
    const float* Wk  = (const float*)d_in[4];
    const float* bk  = (const float*)d_in[5];
    const float* Wv  = (const float*)d_in[6];
    const float* bv  = (const float*)d_in[7];
    const float* Wp1 = (const float*)d_in[8];
    const float* p_g = (const float*)d_in[9];
    const float* p_b = (const float*)d_in[10];
    const float* p_m = (const float*)d_in[11];
    const float* p_v = (const float*)d_in[12];
    const float* Wp2 = (const float*)d_in[13];
    const float* bp2 = (const float*)d_in[14];
    const float* w1_g = (const float*)d_in[15];
    const float* w1_b = (const float*)d_in[16];
    const float* w1_m = (const float*)d_in[17];
    const float* w1_v = (const float*)d_in[18];
    const float* Ww1  = (const float*)d_in[19];
    const float* w2_g = (const float*)d_in[20];
    const float* w2_b = (const float*)d_in[21];
    const float* w2_m = (const float*)d_in[22];
    const float* w2_v = (const float*)d_in[23];
    const float* Ww2  = (const float*)d_in[24];
    const float* bw2  = (const float*)d_in[25];

    float* ws   = (float*)d_ws;
    float* q_ws = ws;                          // 65536 * 16 f32 ( 4 MB)
    float* k_ws = ws + (size_t)BN * 16;        // 65536 * 16 f32 ( 4 MB)
    float* v_ws = ws + (size_t)BN * 32;        // 65536 * 64 f32 (16 MB)

    // 128 blocks x 128 threads = 512 waves; 4096 M-tiles / 512 = 8 uniform trips
    qkv_wmma_kernel<<<128, 128, 0, stream>>>(feat, Wq, bq, Wk, bk, Wv, bv,
                                             q_ws, k_ws, v_ws);

    // one thread per point: 65536 / 256 = 256 blocks exactly
    pt_attn_kernel<<<BN / 256, 256, 0, stream>>>(points, q_ws, k_ws, v_ws,
        Wp1, p_g, p_b, p_m, p_v, Wp2, bp2,
        w1_g, w1_b, w1_m, w1_v, Ww1,
        w2_g, w2_b, w2_m, w2_v, Ww2, bw2,
        (float*)d_out);
}
